// QAConvLoss_19370302505055
// MI455X (gfx1250) — matrix operand order repaired and checked by reading, not compiled, not run on hardware
//
#include <hip/hip_runtime.h>
#include <hip/hip_bf16.h>
#include <math.h>

// Problem constants (from reference)
#define KCLS 128
#define CCH  128
#define BB   32
#define HWSZ 192            // H*W = 24*8
#define THW  384            // 2*HW
#define PITCH 136           // 128 + 8 bf16 pad -> 272B row = 17*16B (aligned, bank-spread)
#define EPSBN 1e-5f

typedef __attribute__((ext_vector_type(16))) __bf16          v16bf;
typedef __attribute__((ext_vector_type(8)))  float           v8f;
typedef __attribute__((ext_vector_type(8)))  unsigned short  v8us;

union FragBF { v16bf v; v8us h[2]; };

__device__ __forceinline__ unsigned short f2bf(float x) {
  unsigned int u = __float_as_uint(x);
  u += 0x7FFFu + ((u >> 16) & 1u);          // round-to-nearest-even
  return (unsigned short)(u >> 16);
}

// -----------------------------------------------------------------------------
// K1: per (k,b): corr = f[b]^T @ m[k]  (192x192x128, bf16 WMMA, f32 acc),
//     dual-axis max-pool -> score[384], reduced to (sum, sumsq, dot(score,w)).
// Each wave owns a 32x192 slab of corr: 2 A-fragment sets stay in registers and
// every B fragment loaded from LDS feeds TWO independent WMMAs (1:1 wmma:ds_load).
// -----------------------------------------------------------------------------
__global__ __launch_bounds__(192) void qaconv_corr_kernel(
    const float* __restrict__ feature,       // [B, C, HW]
    const float* __restrict__ class_memory,  // [K, C, HW]
    const float* __restrict__ fc_w,          // [2*HW]
    float* __restrict__ partial)             // [K*B, 3]
{
  extern __shared__ unsigned char smem_raw[];
  unsigned short* fT = (unsigned short*)smem_raw;        // [192][PITCH] bf16, fT[q][c]
  unsigned short* mT = fT + HWSZ * PITCH;                // [192][PITCH] bf16, mT[p][c]
  float* colmax_w = (float*)(mT + HWSZ * PITCH);         // [6][192]
  float* score    = colmax_w + 6 * HWSZ;                 // [384]
  float* red      = score + THW;                         // [3][192]

  const int tid  = threadIdx.x;            // 0..191
  const int kb   = blockIdx.x;             // 0..4095
  const int k    = kb >> 5;
  const int b    = kb & 31;
  const int lane = tid & 31;
  const int wv   = tid >> 5;               // 0..5
  const int n16  = lane & 15;
  const int g    = lane >> 4;              // half-wave group

  // ---- stage inputs to LDS transposed to [pos][c], fp32 -> bf16 (packed b32) ----
  const float* fb = feature      + (size_t)b * (CCH * HWSZ);
  const float* mk = class_memory + (size_t)k * (CCH * HWSZ);
  #pragma unroll 4
  for (int c = 0; c < CCH; c += 2) {
    __builtin_prefetch(&mk[(c + 8 < CCH ? c + 8 : c) * HWSZ + tid], 0, 3);
    float f0 = fb[c * HWSZ + tid], f1 = fb[(c + 1) * HWSZ + tid];   // q = tid
    float m0 = mk[c * HWSZ + tid], m1 = mk[(c + 1) * HWSZ + tid];   // p = tid
    unsigned int pf = (unsigned int)f2bf(f0) | ((unsigned int)f2bf(f1) << 16);
    unsigned int pm = (unsigned int)f2bf(m0) | ((unsigned int)f2bf(m1) << 16);
    *(unsigned int*)(fT + (size_t)tid * PITCH + c) = pf;
    *(unsigned int*)(mT + (size_t)tid * PITCH + c) = pm;
  }
  __syncthreads();

  const float NEGINF = -3.0e38f;
  const int qt0 = wv * 2, qt1 = qt0 + 1;    // two exclusive q-tile rows per wave
  const int q0  = qt0 * 16 + n16;
  const unsigned short* arow0 = fT + (size_t)q0 * PITCH;
  const unsigned short* arow1 = arow0 + 16 * PITCH;

  // A fragments (16x32 bf16): lanes 0-15 hold K {c0..c0+7, c0+16..c0+23},
  // lanes 16-31 hold K {c0+8..c0+15, c0+24..c0+31}  -> two 16B LDS chunks.
  FragBF afr0[4], afr1[4];
  #pragma unroll
  for (int s = 0; s < 4; ++s) {
    const int c0 = 32 * s + 8 * g;
    afr0[s].h[0] = *(const v8us*)(arow0 + c0);
    afr0[s].h[1] = *(const v8us*)(arow0 + c0 + 16);
    afr1[s].h[0] = *(const v8us*)(arow1 + c0);
    afr1[s].h[1] = *(const v8us*)(arow1 + c0 + 16);
  }

  float cmreg[12];                          // per-lane col-max partial (rows r+8g, both q-tiles)
  float rm0[8], rm1[8];                     // per-lane row-max, rows r + 8g
  #pragma unroll
  for (int pt = 0; pt < 12; ++pt) cmreg[pt] = NEGINF;
  #pragma unroll
  for (int r = 0; r < 8; ++r) { rm0[r] = NEGINF; rm1[r] = NEGINF; }

  #pragma unroll
  for (int pt = 0; pt < 12; ++pt) {         // fully unrolled: independent chains
    const unsigned short* brow = mT + (size_t)(pt * 16 + n16) * PITCH;
    v8f acc0 = {}, acc1 = {};
    #pragma unroll
    for (int s = 0; s < 4; ++s) {
      // B fragment (32x16 bf16): col = n16, K = {8g..8g+7, 16+8g..16+8g+7} (same as A) -- loaded
      // ONCE and consumed by two independent WMMAs.
      FragBF bfr;
      const int c0 = 32 * s + 8 * g;
      bfr.h[0] = *(const v8us*)(brow + c0);
      bfr.h[1] = *(const v8us*)(brow + c0 + 16);
      acc0 = __builtin_amdgcn_wmma_f32_16x16x32_bf16(
                 false, afr0[s].v, false, bfr.v, (short)0, acc0, false, false);
      acc1 = __builtin_amdgcn_wmma_f32_16x16x32_bf16(
                 false, afr1[s].v, false, bfr.v, (short)0, acc1, false, false);
    }
    // per-column partial (max over this half-wave's rows in both q-tiles)
    float cm = fmaxf(acc0[0], acc1[0]);
    #pragma unroll
    for (int r = 1; r < 8; ++r) cm = fmaxf(cm, fmaxf(acc0[r], acc1[r]));
    cmreg[pt] = fmaxf(cmreg[pt], cm);
    // per-row (max over p): accumulate per-lane, butterfly after the p loop
    #pragma unroll
    for (int r = 0; r < 8; ++r) {
      rm0[r] = fmaxf(rm0[r], acc0[r]);
      rm1[r] = fmaxf(rm1[r], acc1[r]);
    }
  }

  // reduce row maxima across the 16 lanes of each half-wave
  #pragma unroll
  for (int off = 1; off < 16; off <<= 1) {
    #pragma unroll
    for (int r = 0; r < 8; ++r) {
      rm0[r] = fmaxf(rm0[r], __shfl_xor(rm0[r], off, 32));
      rm1[r] = fmaxf(rm1[r], __shfl_xor(rm1[r], off, 32));
    }
  }
  // rows are exclusively owned by this wave: direct write (axis=3 half)
  #pragma unroll
  for (int r = 0; r < 8; ++r) {
    if (n16 == r) {
      score[HWSZ + qt0 * 16 + 8 * g + r] = rm0[r];
      score[HWSZ + qt1 * 16 + 8 * g + r] = rm1[r];
    }
  }

  // combine the two half-wave row-subsets once, then publish per-wave col maxima
  #pragma unroll
  for (int pt = 0; pt < 12; ++pt) {
    float cm = fmaxf(cmreg[pt], __shfl_xor(cmreg[pt], 16, 32));
    if (g == 0) colmax_w[wv * HWSZ + pt * 16 + n16] = cm;
  }
  __syncthreads();
  {
    float v = colmax_w[tid];
    #pragma unroll
    for (int w2 = 1; w2 < 6; ++w2) v = fmaxf(v, colmax_w[w2 * HWSZ + tid]);
    score[tid] = v;                         // axis=2 half
  }
  __syncthreads();

  // BN+FC are affine => only (sum, sumsq, dot(score, w)) are needed downstream
  {
    float s1 = score[tid], s2 = score[tid + HWSZ];
    float w1 = fc_w[tid],  w2 = fc_w[tid + HWSZ];
    red[tid]            = s1 + s2;
    red[HWSZ + tid]     = s1 * s1 + s2 * s2;
    red[2 * HWSZ + tid] = s1 * w1 + s2 * w2;
  }
  __syncthreads();
  for (int st = 96; st >= 6; st >>= 1) {    // 192 -> 96 -> ... -> 6
    if (tid < st) {
      red[tid]            += red[tid + st];
      red[HWSZ + tid]     += red[HWSZ + tid + st];
      red[2 * HWSZ + tid] += red[2 * HWSZ + tid + st];
    }
    __syncthreads();
  }
  if (tid == 0) {
    float s = 0.f, q = 0.f, d = 0.f;
    #pragma unroll
    for (int i = 0; i < 6; ++i) {
      s += red[i]; q += red[HWSZ + i]; d += red[2 * HWSZ + i];
    }
    partial[kb * 3 + 0] = s;
    partial[kb * 3 + 1] = q;
    partial[kb * 3 + 2] = d;
  }
}

// -----------------------------------------------------------------------------
// K2: global BN #1 -> logits -> global BN #2 -> focal BCE + argmax accuracy.
// One block, 1024 threads; one wave32 per batch sample for the final stage.
// -----------------------------------------------------------------------------
__global__ __launch_bounds__(1024) void qaconv_loss_kernel(
    const float* __restrict__ partial,      // [4096][3]
    const float* __restrict__ fc_w, const float* __restrict__ fc_b,
    const float* __restrict__ bn_gamma, const float* __restrict__ bn_beta,
    const float* __restrict__ logit_gamma, const float* __restrict__ logit_beta,
    const int* __restrict__ target,
    float* __restrict__ out)                // [64]: loss[32] ++ acc[32]
{
  __shared__ float sred[1024];
  __shared__ float slog[BB * KCLS];         // raw logits as [b][k]
  const int tid = threadIdx.x;

  auto block_sum = [&](float v) -> float {
    __syncthreads();
    sred[tid] = v; __syncthreads();
    for (int st = 512; st > 0; st >>= 1) {
      if (tid < st) sred[tid] += sred[tid + st];
      __syncthreads();
    }
    return sred[0];
  };

  float s = 0.f, q = 0.f;
  #pragma unroll
  for (int i = 0; i < 4; ++i) {
    int r = tid + i * 1024;
    s += partial[r * 3 + 0];
    q += partial[r * 3 + 1];
  }
  float S  = block_sum(s);
  float Q  = block_sum(q);
  float pw = (tid < THW) ? fc_w[tid] : 0.f;
  float SW = block_sum(pw);

  const float N1 = (float)(KCLS * BB * THW);
  float mean1 = S / N1;
  float var1  = Q / N1 - mean1 * mean1;     // biased variance, as torch BN
  float a1 = rsqrtf(var1 + EPSBN) * bn_gamma[0];
  float c1 = (bn_beta[0] - a1 * mean1) * SW + fc_b[0];

  float s2 = 0.f, q2 = 0.f;
  #pragma unroll
  for (int i = 0; i < 4; ++i) {
    int r = tid + i * 1024;                 // r = k*32 + b
    float raw = a1 * partial[r * 3 + 2] + c1;
    slog[(r & 31) * KCLS + (r >> 5)] = raw; // transpose to [b][k]
    s2 += raw; q2 += raw * raw;
  }
  float S2 = block_sum(s2);
  float Q2 = block_sum(q2);
  const float N2 = (float)(KCLS * BB);
  float mean2 = S2 / N2;
  float var2  = Q2 / N2 - mean2 * mean2;
  float a2 = rsqrtf(var2 + EPSBN) * logit_gamma[0];
  float d2 = logit_beta[0] - a2 * mean2;
  __syncthreads();

  // wave32 per batch sample b: focal BCE sum + first-max argmax over K=128
  const int b    = tid >> 5;
  const int lane = tid & 31;
  const int tgt  = target[b];
  float lsum = 0.f, bv = -3.0e38f;
  int bk = 0x7fffffff;
  #pragma unroll
  for (int j = 0; j < 4; ++j) {
    int kk = lane + j * 32;
    float x  = a2 * slog[b * KCLS + kk] + d2;
    float ax = fabsf(x);
    float lse = log1pf(expf(-ax));
    float sp_pos = fmaxf(x, 0.f) + lse;     // softplus(x)
    float sp_neg = fmaxf(-x, 0.f) + lse;    // softplus(-x)
    float prob = 1.f / (1.f + expf(-x));
    bool  y = (kk == tgt);
    float bce = y ? sp_neg : sp_pos;
    float wgt = y ? (1.f - prob) : prob;
    lsum += bce * wgt * wgt;
    if (x > bv) { bv = x; bk = kk; }        // within-lane k ascending -> first max
  }
  #pragma unroll
  for (int off = 16; off > 0; off >>= 1) {
    lsum += __shfl_xor(lsum, off, 32);
    float ov = __shfl_xor(bv, off, 32);
    int   ok = __shfl_xor(bk, off, 32);
    if (ov > bv || (ov == bv && ok < bk)) { bv = ov; bk = ok; }
  }
  if (lane == 0) {
    out[b]      = lsum;
    out[BB + b] = (bk == tgt) ? 1.f : 0.f;
  }
}

// -----------------------------------------------------------------------------
extern "C" void kernel_launch(void* const* d_in, const int* in_sizes, int n_in,
                              void* d_out, int out_size, void* d_ws, size_t ws_size,
                              hipStream_t stream) {
  (void)in_sizes; (void)n_in; (void)out_size; (void)ws_size;
  const float* feature      = (const float*)d_in[0];
  const int*   target       = (const int*)  d_in[1];   // JAX default: int32
  const float* class_memory = (const float*)d_in[2];
  const float* bn_gamma     = (const float*)d_in[3];
  const float* bn_beta      = (const float*)d_in[4];
  const float* fc_w         = (const float*)d_in[5];
  const float* fc_b         = (const float*)d_in[6];
  const float* logit_gamma  = (const float*)d_in[7];
  const float* logit_beta   = (const float*)d_in[8];
  float* out     = (float*)d_out;
  float* partial = (float*)d_ws;          // 4096 * 3 floats = 48 KB scratch

  const size_t smem = (size_t)(2 * HWSZ * PITCH) * sizeof(unsigned short)
                    + (size_t)(6 * HWSZ + THW + 3 * HWSZ) * sizeof(float); // 112,896 B

  qaconv_corr_kernel<<<dim3(KCLS * BB), dim3(192), smem, stream>>>(
      feature, class_memory, fc_w, partial);
  qaconv_loss_kernel<<<dim3(1), dim3(1024), 0, stream>>>(
      partial, fc_w, fc_b, bn_gamma, bn_beta, logit_gamma, logit_beta, target, out);
}
